// CompanySpecificHeads_20590073217613
// MI455X (gfx1250) — matrix operand order repaired and checked by reading, not compiled, not run on hardware
//
#include <hip/hip_runtime.h>
#include <hip/hip_bf16.h>
#include <math.h>

// ---------------------------------------------------------------------------
// CompanySpecificHeads: out[b] = GELU(z[b] @ W1[c_b] + b1[c_b]) @ W2[c_b] + b2[c_b]
// B=4096, C=64, D=512, H=1024, O=1.  Grouped-by-company bf16 WMMA GEMM.
// fp32->bf16 packing via single v_perm_b32; triple-buffered B fragments to
// avoid WMMA->VALU WAR hazard NOPs.
// ---------------------------------------------------------------------------

#define D_DIM  512
#define H_DIM  1024
#define C_DIM  64
#define TILE_M 16
#define SPLIT  4      // workgroups per company

typedef __attribute__((ext_vector_type(16))) __bf16 v16bf;
typedef __attribute__((ext_vector_type(8)))  float  v8f;

// Pack {bf16(lo), bf16(hi)} into one dword with a single v_perm_b32.
__device__ __forceinline__ unsigned pack_bf16x2(float lo, float hi) {
    return __builtin_amdgcn_perm(__float_as_uint(hi), __float_as_uint(lo), 0x07060302u);
}

// ---- pass 1: zero counters (counts[64] ++ cursors[64] contiguous) ----------
__global__ void zero_ws_kernel(int* p) {
    p[threadIdx.x] = 0;
}

// ---- pass 2: per-company histogram ----------------------------------------
__global__ void count_kernel(const int* __restrict__ cid, int* __restrict__ counts, int B) {
    int b = blockIdx.x * blockDim.x + threadIdx.x;
    if (b < B) atomicAdd(&counts[cid[b]], 1);
}

// ---- pass 3: exclusive scan over 64 counts --------------------------------
__global__ void scan_kernel(const int* __restrict__ counts, int* __restrict__ offsets) {
    if (threadIdx.x == 0) {
        int acc = 0;
        for (int i = 0; i < C_DIM; ++i) { offsets[i] = acc; acc += counts[i]; }
        offsets[C_DIM] = acc;
    }
}

// ---- pass 4: scatter sample ids into grouped order ------------------------
__global__ void scatter_kernel(const int* __restrict__ cid, const int* __restrict__ offsets,
                               int* __restrict__ cursors, int* __restrict__ grouped, int B) {
    int b = blockIdx.x * blockDim.x + threadIdx.x;
    if (b < B) {
        int c = cid[b];
        int pos = offsets[c] + atomicAdd(&cursors[c], 1);
        grouped[pos] = b;
    }
}

// ---- pass 5: grouped GEMM + GELU + reduce ---------------------------------
union BFrag { v16bf v; unsigned u[8]; };

__device__ __forceinline__ void fill_bfrag(BFrag& dst, const float* __restrict__ Brow,
                                           int wave, int t) {
    const int n0 = (wave * 8 + t) * 16;
    #pragma unroll
    for (int q = 0; q < 8; ++q) {
        int s0 = 2 * q + ((q >= 4) ? 8 : 0);
        dst.u[q] = pack_bf16x2(Brow[(size_t)s0 * H_DIM + n0],
                               Brow[(size_t)(s0 + 1) * H_DIM + n0]);
    }
}

__global__ __launch_bounds__(256)
void heads_gemm_kernel(const float* __restrict__ z,
                       const int*   __restrict__ offsets,
                       const int*   __restrict__ grouped,
                       const float* __restrict__ W1,
                       const float* __restrict__ b1,
                       const float* __restrict__ W2,
                       const float* __restrict__ b2,
                       float*       __restrict__ out) {
    const int c     = blockIdx.x / SPLIT;
    const int split = blockIdx.x % SPLIT;
    const int tid   = threadIdx.x;
    const int wave  = tid >> 5;     // 8 waves (wave32)
    const int lane  = tid & 31;

    __shared__ unsigned Atile[TILE_M * D_DIM / 2];     // 16 KB bf16 z-tile (2/dword)
    __shared__ float sb1[H_DIM];                       // 4 KB bias1
    __shared__ float sw2[H_DIM];                       // 4 KB W2 column
    __shared__ float rowsum[TILE_M];
    __shared__ int   rowidx[TILE_M];

    for (int i = tid; i < H_DIM; i += 256) {
        sb1[i] = b1[c * H_DIM + i];
        sw2[i] = W2[c * H_DIM + i];
    }

    const int begin  = offsets[c];
    const int end    = offsets[c + 1];
    const int nrows  = end - begin;
    const int ntiles = (nrows + TILE_M - 1) / TILE_M;
    const float bias2 = b2[c];
    const float* W1c  = W1 + (size_t)c * D_DIM * H_DIM;

    for (int tile = split; tile < ntiles; tile += SPLIT) {
        const int row0 = begin + tile * TILE_M;
        if (tid < TILE_M) {
            int r = row0 + tid;
            rowidx[tid] = (r < end) ? grouped[r] : -1;
            rowsum[tid] = 0.0f;
        }
        __syncthreads();

        // stage A tile (16 x 512) as bf16 into LDS, zero-pad invalid rows.
        for (int e = tid; e < TILE_M * D_DIM / 2; e += 256) {
            int row = e >> 8;                 // (e*2)/512
            int col = (e & 255) * 2;
            int g = rowidx[row];
            float2 v = make_float2(0.0f, 0.0f);
            if (g >= 0) v = *(const float2*)&z[(size_t)g * D_DIM + col];
            Atile[e] = pack_bf16x2(v.x, v.y);
        }
        __syncthreads();

        // 8 waves x 8 n-tiles x 16 cols = 1024 = H
        v8f acc[8] = {};

        const int rowA  = lane & 15;
        const int halfA = lane >> 4;
        const unsigned short* AtileS = (const unsigned short*)Atile;
        for (int kb = 0; kb < D_DIM; kb += 32) {
            // A fragment (16x32 bf16): lane=M, per-lane contiguous -> 2x ds_load_b128
            union { v16bf v; unsigned short s[16]; } afrag;
            #pragma unroll
            for (int v = 0; v < 8; ++v) {
                int k = kb + ((v >= 4) ? 16 : 0) + halfA * 8 + (v & 3) * 2;
                const unsigned short* p = &AtileS[rowA * D_DIM + k];
                afrag.s[2 * v]     = p[0];
                afrag.s[2 * v + 1] = p[1];
            }
            // B fragments (32x16 bf16): lane%16 = N, elements = K in table order.
            // Triple-buffered, filled one tile ahead of the consuming WMMA so
            // no VALU write lands on WMMA-read VGPRs within the hazard window.
            const float* Brow = W1c + (size_t)(kb + (lane >> 4) * 8) * H_DIM + (lane & 15);
            BFrag bfrag[3];
            fill_bfrag(bfrag[0], Brow, wave, 0);
            #pragma unroll
            for (int t = 0; t < 8; ++t) {
                if (t < 7) fill_bfrag(bfrag[(t + 1) % 3], Brow, wave, t + 1);
                acc[t] = __builtin_amdgcn_wmma_f32_16x16x32_bf16(
                    false, afrag.v, false, bfrag[t % 3].v, (short)0, acc[t],
                    false, false);
            }
        }

        // epilogue: +b1, exact GELU, dot with W2, per-row LDS reduction
        const int ncol = lane & 15;
        #pragma unroll
        for (int r = 0; r < 8; ++r) {
            int m = (lane < 16) ? r : (r + 8);   // C/D layout: VGPR r -> M=r / M=r+8
            float partial = 0.0f;
            #pragma unroll
            for (int t = 0; t < 8; ++t) {
                int h = (wave * 8 + t) * 16 + ncol;
                float x = acc[t][r] + sb1[h];
                float g = 0.5f * x * (1.0f + erff(x * 0.70710678118654752f));
                partial += g * sw2[h];
            }
            atomicAdd(&rowsum[m], partial);
        }
        __syncthreads();

        if (tid < TILE_M) {
            int g = rowidx[tid];
            if (g >= 0) out[g] = rowsum[tid] + bias2;
        }
        __syncthreads();
    }
}

// ---------------------------------------------------------------------------
extern "C" void kernel_launch(void* const* d_in, const int* in_sizes, int n_in,
                              void* d_out, int out_size, void* d_ws, size_t ws_size,
                              hipStream_t stream) {
    const float* z   = (const float*)d_in[0];
    const int*   cid = (const int*)  d_in[1];
    const float* W1  = (const float*)d_in[2];
    const float* b1  = (const float*)d_in[3];
    const float* W2  = (const float*)d_in[4];
    const float* b2  = (const float*)d_in[5];
    float* out = (float*)d_out;

    const int B = in_sizes[1];          // company_id is [B]

    int* ws      = (int*)d_ws;
    int* counts  = ws;            // 64
    int* cursors = ws + 64;       // 64
    int* offsets = ws + 128;      // 65
    int* grouped = ws + 256;      // B

    zero_ws_kernel<<<1, 128, 0, stream>>>(counts);            // counts + cursors
    count_kernel<<<(B + 255) / 256, 256, 0, stream>>>(cid, counts, B);
    scan_kernel<<<1, 1, 0, stream>>>(counts, offsets);
    scatter_kernel<<<(B + 255) / 256, 256, 0, stream>>>(cid, offsets, cursors, grouped, B);
    heads_gemm_kernel<<<C_DIM * SPLIT, 256, 0, stream>>>(z, offsets, grouped,
                                                         W1, b1, W2, b2, out);
}
